// QRNNCell__31997506355210
// MI455X (gfx1250) — matrix operand order repaired and checked by reading, not compiled, not run on hardware
//
#include <hip/hip_runtime.h>
#include <hip/hip_bf16.h>

// ---------------------------------------------------------------------------
// QRNN (fo-pool, window=2) for MI455X / gfx1250, wave32.
//   B=32, T=2048, F=400, H=1152.  M=B*T=65536, K=2F=800, N=3H=3456.
// Phase 1: x -> bf16            Phase 2: W -> Wt[3456,800] bf16 (transposed)
// Phase 3: WMMA GEMM (v_wmma_f32_16x16x32_bf16), B tile double-buffered
//          through LDS, B fragments ping-ponged in registers, A fragments
//          ping-ponged across iterations; fused bias/tanh/sigmoid epilogue.
// Phase 4: sequential scan over T per (b,h): h = zf + fm*h; out = o*h.
// ---------------------------------------------------------------------------

#define QB   32
#define QT   2048
#define QF   400
#define QH   1152
#define QM   (QB * QT)        // 65536 rows
#define QK   (2 * QF)         // 800
#define QN3  (3 * QH)         // 3456

typedef __bf16 bf16;
typedef bf16  v16bf __attribute__((ext_vector_type(16)));
typedef float v8f   __attribute__((ext_vector_type(8)));

union FragU { uint4 u[2]; v16bf v; };

__device__ __forceinline__ float fast_sigmoid(float x) {
  return __builtin_amdgcn_rcpf(1.0f + __expf(-x));
}
__device__ __forceinline__ float fast_tanh(float x) {
  // tanh(x) = 2*sigmoid(2x) - 1
  return fmaf(2.0f, fast_sigmoid(2.0f * x), -1.0f);
}

// ---------------------------------------------------------------- phase 1
__global__ void qrnn_cvt_x(const float* __restrict__ x, bf16* __restrict__ xh, int n) {
  int i = blockIdx.x * blockDim.x + threadIdx.x;
  if (i < n) xh[i] = (bf16)x[i];
}

// ---------------------------------------------------------------- phase 2
// Wt[n*QK + k] = W[k*QN3 + n]   (bf16)
__global__ void qrnn_transpose_w(const float* __restrict__ W, bf16* __restrict__ Wt, int n) {
  int i = blockIdx.x * blockDim.x + threadIdx.x;
  if (i >= n) return;
  int k = i % QK;
  int c = i / QK;
  Wt[i] = (bf16)W[(size_t)k * QN3 + c];
}

// ---------------------------------------------------------------- phase 3
// grid = (QM/128, QH/64), block = 256 (8 waves). Wave w of block bx owns
// rows m0 = (bx*8+w)*16; all waves share columns h0..h0+63 of all 3 gates.
__global__ void __launch_bounds__(256)
qrnn_gemm_wmma(const bf16* __restrict__ xh,   // [QM, QF] bf16 (x cast)
               const bf16* __restrict__ Wt,   // [QN3, QK] bf16 (W transposed)
               const float* __restrict__ bias,// [QN3]
               float* __restrict__ zfb,       // [QM, QH]
               float* __restrict__ fmb,       // [QM, QH]
               float* __restrict__ out)       // [QM, QH]  (sigmoid(o) staged)
{
  // LDS B tile: [buf][frag(12)][chunk(2)][lane(32)] of 16B => 2 x 12KB.
  // Chunk-plane-major: a wave-wide ds_load_b128 reads 512 contiguous bytes.
  __shared__ uint4 ldsb[2][12 * 2 * 32];

  const int tid  = threadIdx.x;
  const int lane = tid & 31;
  const int wave = tid >> 5;
  const int m0   = (blockIdx.x * 8 + wave) * 16;
  const int h0   = blockIdx.y * 64;

  const int laneLo = lane & 15;          // row (A) / column (B) within tile
  const int hiHalf = lane >> 4;          // 0: lanes 0-15, 1: lanes 16-31

  // ---- A-fragment addressing (16-bit A 16x32 layout, ISA 7.12.2) ----
  // lanes 0-15 : row M=lane,    K = {kk..kk+7, kk+16..kk+23}
  // lanes 16-31: row M=lane-16, K = {kk+8..kk+15, kk+24..kk+31}
  // Single-base indexing (no pointer select) so these stay global_load_b128.
  const int rowM  = m0 + laneLo;
  const int trow  = rowM & (QT - 1);     // T = 2048 is a power of two
  const bool t0   = (trow == 0);
  const int koffA = hiHalf ? 8 : 0;
  const uint4 z4 = make_uint4(0u, 0u, 0u, 0u);

  auto loadA = [&](int kk) -> FragU {
    FragU a;
#pragma unroll
    for (int half = 0; half < 2; ++half) {
      const int ks = kk + koffA + 16 * half;
      const bool prev = (ks >= QF);                 // wave-uniform per step
      const int sub   = (prev && !t0) ? 1 : 0;      // shift to row m-1
      const size_t idx = (size_t)(rowM - sub) * QF + (ks - (prev ? QF : 0));
      uint4 v = *reinterpret_cast<const uint4*>(xh + idx);
      if (prev && t0) v = z4;                       // x_{-1} = 0
      a.u[half] = v;
    }
    return a;
  };

  // ---- cooperative B-tile fill: 768 16B chunks, 3 per thread ----
  // chunk q = frag*64 + chunk*32 + l.  (16-bit B 32x16 layout:
  //  lane n: col n, K={kk..kk+7, kk+16..kk+23} ; lane 16+n: col n, K={kk+8..kk+15, kk+24..kk+31})
  const bf16* srcB[3];
  int dstQ[3];
#pragma unroll
  for (int s = 0; s < 3; ++s) {
    int q = tid + s * 256;
    int fi = q >> 6;            // fragment 0..11  (g = fi>>2, j = fi&3)
    int c  = (q >> 5) & 1;      // 16B chunk within fragment
    int l  = q & 31;            // lane the chunk feeds
    int col   = (fi >> 2) * QH + h0 + (fi & 3) * 16 + (l & 15);
    int koffB = ((l & 16) ? 8 : 0) + 16 * c;
    srcB[s] = Wt + (size_t)col * QK + koffB;
    dstQ[s] = q;
  }

  v8f acc[12];
#pragma unroll
  for (int i = 0; i < 12; ++i) acc[i] = v8f{0.f,0.f,0.f,0.f,0.f,0.f,0.f,0.f};

  // B-tile prefetch registers (global -> LDS staging)
  uint4 pr0, pr1, pr2;

  // ---- prologue: tile 0 into buf 0, A fragment 0 prefetched ----
  pr0 = *reinterpret_cast<const uint4*>(srcB[0]);
  pr1 = *reinterpret_cast<const uint4*>(srcB[1]);
  pr2 = *reinterpret_cast<const uint4*>(srcB[2]);
  FragU aA = loadA(0), aB;
  ldsb[0][dstQ[0]] = pr0;
  ldsb[0][dstQ[1]] = pr1;
  ldsb[0][dstQ[2]] = pr2;
  __syncthreads();

  // One K-step (32 wide). aUse feeds this step's WMMAs; aNext is loaded for
  // the following step. B fragments ping-pong between two register sets so
  // each WMMA only waits for dscnt<=2 (latency hidden by the running WMMA).
  auto kstep = [&](int it, FragU& aUse, FragU& aNext) {
    const int p = it & 1;
    const bool more = (it < 24);
    if (more) {
      const int kk = (it + 1) * 32;
      pr0 = *reinterpret_cast<const uint4*>(srcB[0] + kk);
      pr1 = *reinterpret_cast<const uint4*>(srcB[1] + kk);
      pr2 = *reinterpret_cast<const uint4*>(srcB[2] + kk);
      aNext = loadA(kk);
    }
    FragU bfr0, bfr1;
    bfr0.u[0] = ldsb[p][lane];
    bfr0.u[1] = ldsb[p][32 + lane];
#pragma unroll
    for (int i = 0; i < 12; ++i) {
      FragU& bu = (i & 1) ? bfr1 : bfr0;
      FragU& bn = (i & 1) ? bfr0 : bfr1;
      if (i < 11) {
        bn.u[0] = ldsb[p][(i + 1) * 64 + lane];
        bn.u[1] = ldsb[p][(i + 1) * 64 + 32 + lane];
      }
      acc[i] = __builtin_amdgcn_wmma_f32_16x16x32_bf16(
          false, aUse.v, false, bu.v, (short)0, acc[i], false, false);
    }
    if (more) {
      ldsb[p ^ 1][dstQ[0]] = pr0;
      ldsb[p ^ 1][dstQ[1]] = pr1;
      ldsb[p ^ 1][dstQ[2]] = pr2;
    }
    __syncthreads();
  };

  // 25 K-steps; manual 2x unroll alternates aA/aB without register copies.
#pragma unroll 1
  for (int ii = 0; ii < 12; ++ii) {
    kstep(2 * ii,     aA, aB);
    kstep(2 * ii + 1, aB, aA);
  }
  kstep(24, aA, aB);

  // ---- epilogue: bias + activations, fused gate products ----
  // C/D layout: VGPR r, lanes 0-15 -> (M=r, N=lane); lanes 16-31 -> (M=8+r, N=lane-16)
  const int rbase = m0 + (hiHalf << 3);
#pragma unroll
  for (int j = 0; j < 4; ++j) {
    const int hcol = h0 + j * 16 + laneLo;
    const float bz = bias[hcol];
    const float bf = bias[QH + hcol];
    const float bo = bias[2 * QH + hcol];
#pragma unroll
    for (int r = 0; r < 8; ++r) {
      const int row = rbase + r;
      const float zv = acc[j][r]     + bz;
      const float fv = acc[4 + j][r] + bf;
      const float ov = acc[8 + j][r] + bo;
      const float fs = fast_sigmoid(fv);
      const float zs = fast_tanh(zv);
      const float os = fast_sigmoid(ov);
      const size_t o = (size_t)row * QH + hcol;
      zfb[o] = zs * fs;
      fmb[o] = 1.0f - fs;
      out[o] = os;           // staged; scan multiplies by h_t in place
    }
  }
}

// ---------------------------------------------------------------- phase 4
__global__ void __launch_bounds__(256)
qrnn_scan(const float* __restrict__ zfb, const float* __restrict__ fmb,
          const float* __restrict__ h0v, float* __restrict__ out)
{
  const int tid = blockIdx.x * blockDim.x + threadIdx.x;   // 0 .. B*H-1
  const int h = tid % QH;
  const int b = tid / QH;
  float hs = h0v[(size_t)b * QH + h];
  const size_t base = ((size_t)b * QT) * QH + h;

  for (int t = 0; t < QT; t += 8) {
    float a[8], m[8], o[8];
#pragma unroll
    for (int u = 0; u < 8; ++u) {
      const size_t idx = base + (size_t)(t + u) * QH;
      a[u] = zfb[idx];
      m[u] = fmb[idx];
      o[u] = out[idx];
    }
#pragma unroll
    for (int u = 0; u < 8; ++u) {
      hs = fmaf(m[u], hs, a[u]);
      const size_t idx = base + (size_t)(t + u) * QH;
      out[idx] = o[u] * hs;
    }
  }
}

// ---------------------------------------------------------------------------
extern "C" void kernel_launch(void* const* d_in, const int* in_sizes, int n_in,
                              void* d_out, int out_size, void* d_ws, size_t ws_size,
                              hipStream_t stream) {
  const float* x  = (const float*)d_in[0];   // [B,T,F]
  const float* W  = (const float*)d_in[1];   // [2F, 3H]
  const float* bv = (const float*)d_in[2];   // [3H]
  const float* h0 = (const float*)d_in[3];   // [B,H]
  float* out = (float*)d_out;                // [B,T,H]

  // workspace: zf (302MB) | fm (302MB) | xh bf16 (50MB) | Wt bf16 (5.5MB)
  const size_t MH = (size_t)QM * QH;
  float* zfb = (float*)d_ws;
  float* fmb = zfb + MH;
  bf16*  xh  = (bf16*)(fmb + MH);
  bf16*  Wt  = xh + (size_t)QM * QF;

  { // phase 1: x -> bf16  (26,214,400 elements)
    const int n = QM * QF;
    qrnn_cvt_x<<<(n + 255) / 256, 256, 0, stream>>>(x, xh, n);
  }
  { // phase 2: W -> Wt  (2,764,800 elements)
    const int n = QN3 * QK;
    qrnn_transpose_w<<<(n + 255) / 256, 256, 0, stream>>>(W, Wt, n);
  }
  { // phase 3: WMMA GEMM + fused activations
    dim3 grid(QM / 128, QH / 64);   // (512, 18)
    qrnn_gemm_wmma<<<grid, 256, 0, stream>>>(xh, Wt, bv, zfb, fmb, out);
  }
  { // phase 4: recurrence
    const int n = QB * QH;          // 36864
    qrnn_scan<<<n / 256, 256, 0, stream>>>(zfb, fmb, h0, out);
  }
}
